// MultiHeadAttention_50405736186343
// MI455X (gfx1250) — matrix operand order repaired and checked by reading, not compiled, not run on hardware
//
#include <hip/hip_runtime.h>

#define BATCH  4
#define NSEQ   2048
#define DIMSZ  1024
#define NHEADS 16
#define DH     64
#define NTOK   (BATCH * NSEQ)   // 8192
#define LP     40                // proj LDS row stride (80B: 16B-aligned, conflict-free)
#define LPO    32                // out-LN LDS row stride (64B: 16B-aligned, fits 2 buffers)

typedef __bf16 bf16_t;
typedef __attribute__((ext_vector_type(16))) __bf16 v16bf;
typedef __attribute__((ext_vector_type(8)))  float  v8f;

union Frag {
    v16bf v;
    unsigned int u[8];
};

__device__ inline v8f zero8() {
    v8f z;
    #pragma unroll
    for (int i = 0; i < 8; ++i) z[i] = 0.0f;
    return z;
}

// A-fragment (16x32 bf16): lane holds row M=lane%16; VGPR v holds K pair at:
__device__ inline int ka_idx(int v, int hl) { return 2 * v + 16 * hl; }
// B-fragment (32x16 bf16): lane holds col N=lane%16; VGPR v holds K pair at:
__device__ inline int kb_idx(int v, int hl) { return 2 * v + 16 * hl; }

__device__ inline v8f wmma_bf16(v16bf a, v16bf b, v8f c) {
    return __builtin_amdgcn_wmma_f32_16x16x32_bf16(false, a, false, b, (short)0, c, false, false);
}

__device__ inline unsigned int pack_bf16(float lo, float hi) {
    union { bf16_t h[2]; unsigned int u; } p;
    p.h[0] = (bf16_t)lo;
    p.h[1] = (bf16_t)hi;
    return p.u;
}

// CDNA5 async LDS fill: LDS[lds_off..+15] = MEM[gaddr..+15], tracked by ASYNCcnt.
__device__ inline void async_b128(unsigned lds_off, const void* gptr) {
    asm volatile("global_load_async_to_lds_b128 %0, %1, off"
                 :: "v"(lds_off), "v"((unsigned long long)(uintptr_t)gptr)
                 : "memory");
}
// wait until at most `n` async ops remain in flight (per-wave, in-order retire)
#define WAIT_ASYNCCNT(n) asm volatile("s_wait_asynccnt " #n ::: "memory")

// generic->LDS offset (low 32 bits of the LDS-aperture flat address)
__device__ inline unsigned lds_off_of(const void* p) {
    return (unsigned)(uintptr_t)p;
}

// ---------------------------------------------------------------------------
// Kernel 0a: f32 -> bf16 for the three activation tensors (one streaming pass)
// ---------------------------------------------------------------------------
__global__ __launch_bounds__(256) void cvt_act(const float* __restrict__ a,
                                               const float* __restrict__ b,
                                               const float* __restrict__ c,
                                               bf16_t* __restrict__ ab,
                                               bf16_t* __restrict__ bb,
                                               bf16_t* __restrict__ cb)
{
    size_t i = ((size_t)blockIdx.x * 256 + threadIdx.x) * 4;
    float4 x = *(const float4*)(a + i);
    float4 y = *(const float4*)(b + i);
    float4 z = *(const float4*)(c + i);
    uint2 px; px.x = pack_bf16(x.x, x.y); px.y = pack_bf16(x.z, x.w);
    uint2 py; py.x = pack_bf16(y.x, y.y); py.y = pack_bf16(y.z, y.w);
    uint2 pz; pz.x = pack_bf16(z.x, z.y); pz.y = pack_bf16(z.z, z.w);
    *(uint2*)(ab + i) = px;
    *(uint2*)(bb + i) = py;
    *(uint2*)(cb + i) = pz;
}

// ---------------------------------------------------------------------------
// Kernel 0b: weight f32 [k][n] -> bf16 transposed [n][k]
// ---------------------------------------------------------------------------
__global__ __launch_bounds__(256) void cvt_wt(const float* __restrict__ W,
                                              bf16_t* __restrict__ Wt)
{
    int idx = blockIdx.x * 256 + threadIdx.x;   // 1M total
    int k = idx >> 10, n = idx & 1023;
    Wt[(size_t)n * DIMSZ + k] = (bf16_t)W[(size_t)k * DIMSZ + n];
}

// ---------------------------------------------------------------------------
// Kernel 1: projection GEMM  P = X @ W  (bf16, double-buffered async LDS, WMMA)
// mode 0: Q [B,H,N,DH] (scaled)  mode 1: K [B,H,N,DH]  mode 2: Vt [B,H,DH,N]
// Xb: [8192][1024] bf16 row-major.  Wt: [n][k] bf16 (pre-transposed).
// Each wave issues exactly 3 async ops per K-step -> s_wait_asynccnt 3 retires
// the previous buffer while the next tile streams in behind the WMMAs.
// ---------------------------------------------------------------------------
__global__ __launch_bounds__(256) void proj_gemm(const bf16_t* __restrict__ Xb,
                                                 const bf16_t* __restrict__ Wt,
                                                 bf16_t* __restrict__ outp,
                                                 int mode, float scale)
{
    __shared__ __attribute__((aligned(16))) bf16_t sA[2][64 * LP];
    __shared__ __attribute__((aligned(16))) bf16_t sB[2][128 * LP];

    const int tid  = threadIdx.x;
    const int lane = tid & 31;
    const int wv   = tid >> 5;
    const int hl   = lane >> 4;
    const int lr   = lane & 15;
    const int rowBase = blockIdx.x * 64;
    const int colBase = blockIdx.y * 128;
    const int wRow = (wv & 1) * 32;
    const int wCol = (wv >> 1) * 32;

    const int arow = tid >> 2, aoff = (tid & 3) * 16;   // 16B async granules
    const unsigned sA0 = lds_off_of(sA), sB0 = lds_off_of(sB);

    auto issue_tiles = [&](int buf, int k0) {
        const char* gA = (const char*)Xb
                       + ((size_t)(rowBase + arow) * DIMSZ + k0) * 2 + aoff;
        async_b128(sA0 + buf * (64 * LP * 2) + arow * (LP * 2) + aoff, gA);
        #pragma unroll
        for (int i = 0; i < 2; ++i) {
            int t = i * 256 + tid;
            int n = t >> 2, off = (t & 3) * 16;
            const char* gB = (const char*)Wt
                           + ((size_t)(colBase + n) * DIMSZ + k0) * 2 + off;
            async_b128(sB0 + buf * (128 * LP * 2) + n * (LP * 2) + off, gB);
        }
    };

    v8f acc[2][2];
    #pragma unroll
    for (int i = 0; i < 2; ++i)
        #pragma unroll
        for (int j = 0; j < 2; ++j) acc[i][j] = zero8();

    issue_tiles(0, 0);
    int cur = 0;

    #pragma unroll 1
    for (int k0 = 0; k0 < DIMSZ; k0 += 32) {
        if (k0 + 32 < DIMSZ) {
            issue_tiles(cur ^ 1, k0 + 32);
            WAIT_ASYNCCNT(3);           // previous buffer complete; next in flight
        } else {
            WAIT_ASYNCCNT(0);
        }
        __syncthreads();

        const bf16_t* tA = sA[cur];
        const bf16_t* tB = sB[cur];
        Frag a[2], b[2];
        #pragma unroll
        for (int i = 0; i < 2; ++i) {
            int row = wRow + i * 16 + lr;
            #pragma unroll
            for (int v = 0; v < 8; ++v)
                a[i].u[v] = *(const unsigned int*)&tA[row * LP + ka_idx(v, hl)];
        }
        #pragma unroll
        for (int j = 0; j < 2; ++j) {
            int col = wCol + j * 16 + lr;
            #pragma unroll
            for (int v = 0; v < 8; ++v)
                b[j].u[v] = *(const unsigned int*)&tB[col * LP + kb_idx(v, hl)];
        }
        #pragma unroll
        for (int i = 0; i < 2; ++i)
            #pragma unroll
            for (int j = 0; j < 2; ++j)
                acc[i][j] = wmma_bf16(a[i].v, b[j].v, acc[i][j]);
        __syncthreads();                // reads done before next issue overwrites
        cur ^= 1;
    }

    // epilogue: scatter bf16 into head-split layout
    #pragma unroll
    for (int i = 0; i < 2; ++i)
        #pragma unroll
        for (int j = 0; j < 2; ++j) {
            int col = colBase + wCol + j * 16 + lr;
            int h = col >> 6, d = col & 63;
            #pragma unroll
            for (int v = 0; v < 8; ++v) {
                int row = rowBase + wRow + i * 16 + v + 8 * hl;
                int bb = row >> 11;          // / NSEQ
                int n  = row & (NSEQ - 1);
                float val = acc[i][j][v] * scale;
                size_t o;
                if (mode == 2) o = ((size_t)(bb * NHEADS + h) * DH + d) * NSEQ + n;
                else           o = ((size_t)(bb * NHEADS + h) * NSEQ + n) * DH + d;
                outp[o] = (bf16_t)val;
            }
        }
}

// ---------------------------------------------------------------------------
// Kernel 2: flash attention; 256 threads = 8 independent waves, one
// (b,h,16-query block) per wave.  S^T = K @ Q^T (per-lane online softmax),
// O^T += V^T @ P via WMMA.  Q pre-scaled by 1/sqrt(DH).
// ---------------------------------------------------------------------------
__global__ __launch_bounds__(256) void attn_kernel(const bf16_t* __restrict__ Q,
                                                   const bf16_t* __restrict__ K,
                                                   const bf16_t* __restrict__ Vt,
                                                   bf16_t* __restrict__ O)
{
    const int tid  = threadIdx.x;
    const int lane = tid & 31;
    const int wv   = tid >> 5;
    const int hl   = lane >> 4;
    const int lr   = lane & 15;
    const int bh   = blockIdx.y;
    const int b    = bh >> 4;
    const int h    = bh & 15;
    const int qBase = (blockIdx.x * 8 + wv) * 16;

    const bf16_t* Qbh = Q  + (size_t)bh * NSEQ * DH;
    const bf16_t* Kbh = K  + (size_t)bh * NSEQ * DH;
    const bf16_t* Vbh = Vt + (size_t)bh * DH * NSEQ;

    // Q^T B-fragments for 2 K-dim chunks of 32 (Dh = 64)
    Frag bq[2];
    {
        int q = qBase + lr;
        #pragma unroll
        for (int dc = 0; dc < 2; ++dc)
            #pragma unroll
            for (int v = 0; v < 8; ++v) {
                int d = dc * 32 + kb_idx(v, hl);
                bq[dc].u[v] = *(const unsigned int*)&Qbh[(size_t)q * DH + d];
            }
    }

    v8f accO[4];
    #pragma unroll
    for (int dc = 0; dc < 4; ++dc) accO[dc] = zero8();
    float m_run = -3.0e38f;
    float l_run = 0.0f;

    #pragma unroll 1
    for (int j = 0; j < NSEQ; j += 32) {
        // prefetch next K/V tiles into cache (global_prefetch_b8)
        if (j + 32 < NSEQ) {
            __builtin_prefetch(&Kbh[(size_t)(j + 32 + lr) * DH], 0, 3);
            __builtin_prefetch(&Vbh[(size_t)lr * NSEQ + j + 32], 0, 3);
        }

        // --- S^T tiles for keys j..j+15 (t0) and j+16..j+31 (t1) ---
        v8f t0 = zero8(), t1 = zero8();
        #pragma unroll
        for (int dc = 0; dc < 2; ++dc) {
            Frag ak0, ak1;
            #pragma unroll
            for (int v = 0; v < 8; ++v) {
                int d = dc * 32 + ka_idx(v, hl);
                ak0.u[v] = *(const unsigned int*)&Kbh[(size_t)(j + lr) * DH + d];
                ak1.u[v] = *(const unsigned int*)&Kbh[(size_t)(j + 16 + lr) * DH + d];
            }
            t0 = wmma_bf16(ak0.v, bq[dc].v, t0);
            t1 = wmma_bf16(ak1.v, bq[dc].v, t1);
        }

        // --- online softmax (per-lane; lane pair l,l+16 shares the query) ---
        float bm = t0[0];
        #pragma unroll
        for (int v = 1; v < 8; ++v) bm = fmaxf(bm, t0[v]);
        #pragma unroll
        for (int v = 0; v < 8; ++v) bm = fmaxf(bm, t1[v]);
        bm = fmaxf(bm, __shfl_xor(bm, 16, 32));
        float m_new = fmaxf(m_run, bm);
        float corr  = __expf(m_run - m_new);

        float e0[8], e1[8], ls = 0.0f;
        #pragma unroll
        for (int v = 0; v < 8; ++v) { e0[v] = __expf(t0[v] - m_new); ls += e0[v]; }
        #pragma unroll
        for (int v = 0; v < 8; ++v) { e1[v] = __expf(t1[v] - m_new); ls += e1[v]; }
        l_run = l_run * corr + ls;
        m_run = m_new;
        #pragma unroll
        for (int dc = 0; dc < 4; ++dc) accO[dc] = accO[dc] * corr;

        // --- repack P (32 keys x 16 queries) into a B-fragment ---
        float p0[8], p1[8];
        #pragma unroll
        for (int v = 0; v < 8; ++v) {
            p0[v] = __shfl_xor(e0[v], 16, 32);
            p1[v] = __shfl_xor(e1[v], 16, 32);
        }
        Frag bp;
        #pragma unroll
        for (int v = 0; v < 8; ++v) {
            float lo, hi;
            if (hl == 0) {
                if (v < 4) { lo = e0[2 * v];     hi = e0[2 * v + 1]; }
                else       { lo = p0[2 * v - 8]; hi = p0[2 * v - 7]; }
            } else {
                if (v < 4) { lo = p1[2 * v];     hi = p1[2 * v + 1]; }
                else       { lo = e1[2 * v - 8]; hi = e1[2 * v - 7]; }
            }
            bp.u[v] = pack_bf16(lo, hi);
        }

        // --- O^T += V^T @ P  (4 d-chunks of 16) ---
        #pragma unroll
        for (int dc = 0; dc < 4; ++dc) {
            Frag av;
            int d = dc * 16 + lr;
            #pragma unroll
            for (int v = 0; v < 8; ++v)
                av.u[v] = *(const unsigned int*)&Vbh[(size_t)d * NSEQ + j + ka_idx(v, hl)];
            accO[dc] = wmma_bf16(av.v, bp.v, accO[dc]);
        }
    }

    // finalize: divide by full row sum, write O[b, n, h*64+d] bf16
    float lt  = l_run + __shfl_xor(l_run, 16, 32);
    float inv = 1.0f / lt;
    int q = qBase + lr;
    size_t rowO = ((size_t)b * NSEQ + q) * (size_t)(NHEADS * DH);
    #pragma unroll
    for (int dc = 0; dc < 4; ++dc)
        #pragma unroll
        for (int v = 0; v < 8; ++v) {
            int d = dc * 16 + v + 8 * hl;
            O[rowO + h * DH + d] = (bf16_t)(accO[dc][v] * inv);
        }
}

// ---------------------------------------------------------------------------
// Kernel 3: out = LayerNorm(O @ Wo + bo) * gamma + beta + query (residual)
// One block owns 16 full rows; 8 waves sweep 1024 cols; double-buffered async
// bf16 LDS fills.  Waves 0-1 issue 17 async ops per K-step, waves 2-7 issue 16.
// ---------------------------------------------------------------------------
__global__ __launch_bounds__(256) void out_ln_kernel(const bf16_t* __restrict__ O,
                                                     const bf16_t* __restrict__ Wot,
                                                     const float* __restrict__ bo,
                                                     const float* __restrict__ gamma,
                                                     const float* __restrict__ beta,
                                                     const float* __restrict__ query,
                                                     float* __restrict__ outp)
{
    __shared__ __attribute__((aligned(16))) bf16_t sA[2][16 * LPO];
    __shared__ __attribute__((aligned(16))) bf16_t sB[2][1024 * LPO];
    __shared__ float red1[8][16];
    __shared__ float red2[8][16];
    __shared__ float smu[16];
    __shared__ float srs[16];

    const int tid  = threadIdx.x;
    const int lane = tid & 31;
    const int wv   = tid >> 5;
    const int hl   = lane >> 4;
    const int lr   = lane & 15;
    const int tokBase = blockIdx.x * 16;
    const unsigned sA0 = lds_off_of(sA), sB0 = lds_off_of(sB);

    auto issue_tiles = [&](int buf, int k0) {
        if (tid < 64) {   // waves 0-1: 16 rows x 64B of O
            int r = tid >> 2, off = (tid & 3) * 16;
            const char* g = (const char*)O
                          + ((size_t)(tokBase + r) * DIMSZ + k0) * 2 + off;
            async_b128(sA0 + buf * (16 * LPO * 2) + r * (LPO * 2) + off, g);
        }
        #pragma unroll
        for (int i = 0; i < 16; ++i) {   // 1024 cols x 64B of Wot
            int t = i * 256 + tid;
            int n = t >> 2, off = (t & 3) * 16;
            const char* g = (const char*)Wot
                          + ((size_t)n * DIMSZ + k0) * 2 + off;
            async_b128(sB0 + buf * (1024 * LPO * 2) + n * (LPO * 2) + off, g);
        }
    };

    v8f acc[8];
    #pragma unroll
    for (int t = 0; t < 8; ++t) acc[t] = zero8();

    issue_tiles(0, 0);
    int cur = 0;

    #pragma unroll 1
    for (int k0 = 0; k0 < DIMSZ; k0 += 32) {
        if (k0 + 32 < DIMSZ) {
            issue_tiles(cur ^ 1, k0 + 32);
            if (wv < 2) { WAIT_ASYNCCNT(17); } else { WAIT_ASYNCCNT(16); }
        } else {
            WAIT_ASYNCCNT(0);
        }
        __syncthreads();

        const bf16_t* tA = sA[cur];
        const bf16_t* tB = sB[cur];
        Frag a;
        #pragma unroll
        for (int v = 0; v < 8; ++v)
            a.u[v] = *(const unsigned int*)&tA[lr * LPO + ka_idx(v, hl)];
        #pragma unroll
        for (int t = 0; t < 8; ++t) {
            Frag bfr;
            int col = wv * 128 + t * 16 + lr;
            #pragma unroll
            for (int v = 0; v < 8; ++v)
                bfr.u[v] = *(const unsigned int*)&tB[col * LPO + kb_idx(v, hl)];
            acc[t] = wmma_bf16(a.v, bfr.v, acc[t]);
        }
        __syncthreads();
        cur ^= 1;
    }

    // bias + per-row stats (row r = v + 8*hl)
    float s1[8], s2[8];
    #pragma unroll
    for (int v = 0; v < 8; ++v) { s1[v] = 0.0f; s2[v] = 0.0f; }
    #pragma unroll
    for (int t = 0; t < 8; ++t) {
        int col = wv * 128 + t * 16 + lr;
        float bb = bo[col];
        #pragma unroll
        for (int v = 0; v < 8; ++v) {
            float x = acc[t][v] + bb;
            acc[t][v] = x;
            s1[v] += x;
            s2[v] += x * x;
        }
    }
    #pragma unroll
    for (int v = 0; v < 8; ++v)
        #pragma unroll
        for (int m = 1; m < 16; m <<= 1) {
            s1[v] += __shfl_xor(s1[v], m, 32);
            s2[v] += __shfl_xor(s2[v], m, 32);
        }
    if (lr == 0) {
        #pragma unroll
        for (int v = 0; v < 8; ++v) {
            red1[wv][v + 8 * hl] = s1[v];
            red2[wv][v + 8 * hl] = s2[v];
        }
    }
    __syncthreads();
    if (tid < 16) {
        float a1 = 0.0f, a2 = 0.0f;
        #pragma unroll
        for (int w = 0; w < 8; ++w) { a1 += red1[w][tid]; a2 += red2[w][tid]; }
        float mu  = a1 * (1.0f / 1024.0f);
        float var = a2 * (1.0f / 1024.0f) - mu * mu;
        smu[tid] = mu;
        srs[tid] = rsqrtf(var + 1e-5f);
    }
    __syncthreads();

    #pragma unroll
    for (int t = 0; t < 8; ++t) {
        int col = wv * 128 + t * 16 + lr;
        float g  = gamma[col];
        float bt = beta[col];
        #pragma unroll
        for (int v = 0; v < 8; ++v) {
            int r   = v + 8 * hl;
            int tok = tokBase + r;
            float val = (acc[t][v] - smu[r]) * srs[r] * g + bt
                        + query[(size_t)tok * DIMSZ + col];
            outp[(size_t)tok * DIMSZ + col] = val;
        }
    }
}

// ---------------------------------------------------------------------------
extern "C" void kernel_launch(void* const* d_in, const int* in_sizes, int n_in,
                              void* d_out, int out_size, void* d_ws, size_t ws_size,
                              hipStream_t stream)
{
    const float* query = (const float*)d_in[0];
    const float* key_  = (const float*)d_in[1];
    const float* value = (const float*)d_in[2];
    const float* Wq    = (const float*)d_in[3];
    const float* Wk    = (const float*)d_in[4];
    const float* Wv    = (const float*)d_in[5];
    const float* Wo    = (const float*)d_in[6];
    const float* bo    = (const float*)d_in[7];
    const float* gamma = (const float*)d_in[8];
    const float* beta  = (const float*)d_in[9];
    float* out = (float*)d_out;

    const size_t actE = (size_t)NTOK * DIMSZ;     // 8,388,608
    const size_t wE   = (size_t)DIMSZ * DIMSZ;    // 1,048,576
    bf16_t* Xqb = (bf16_t*)d_ws;
    bf16_t* Xkb = Xqb + actE;
    bf16_t* Xvb = Xkb + actE;
    bf16_t* Wqt = Xvb + actE;
    bf16_t* Wkt = Wqt + wE;
    bf16_t* Wvt = Wkt + wE;
    bf16_t* Wot = Wvt + wE;
    bf16_t* Q   = Wot + wE;
    bf16_t* K   = Q   + actE;
    bf16_t* Vt  = K   + actE;
    bf16_t* O   = Vt  + actE;

    cvt_act<<<actE / (256 * 4), 256, 0, stream>>>(query, key_, value, Xqb, Xkb, Xvb);
    cvt_wt<<<wE / 256, 256, 0, stream>>>(Wq, Wqt);
    cvt_wt<<<wE / 256, 256, 0, stream>>>(Wk, Wkt);
    cvt_wt<<<wE / 256, 256, 0, stream>>>(Wv, Wvt);
    cvt_wt<<<wE / 256, 256, 0, stream>>>(Wo, Wot);

    dim3 gProj(NTOK / 64, DIMSZ / 128);
    proj_gemm<<<gProj, 256, 0, stream>>>(Xqb, Wqt, Q,  0, 0.125f); // 1/sqrt(64)
    proj_gemm<<<gProj, 256, 0, stream>>>(Xkb, Wkt, K,  1, 1.0f);
    proj_gemm<<<gProj, 256, 0, stream>>>(Xvb, Wvt, Vt, 2, 1.0f);

    dim3 gAttn(NSEQ / 128, BATCH * NHEADS);
    attn_kernel<<<gAttn, 256, 0, stream>>>(Q, K, Vt, O);

    out_ln_kernel<<<NTOK / 16, 256, 0, stream>>>(O, Wot, bo, gamma, beta, query, out);
}
